// BiLSTM_CRF_68556267979114
// MI455X (gfx1250) — matrix operand order repaired and checked by reading, not compiled, not run on hardware
//
#include <hip/hip_runtime.h>
#include <hip/hip_bf16.h>
#include <stddef.h>

// ---------------------------------------------------------------------------
// BiLSTM-CRF for MI455X (gfx1250, wave32, WMMA).
// Heavy math = bf16 WMMA (v_wmma_f32_16x16x32_bf16), f32 accumulation.
// Weights (8MB bf16) are L2-resident on a 192MB-L2 part -> per-step GEMMs are
// L2-bound; fuse gates into the GEMM step kernel so G never hits HBM.
// ---------------------------------------------------------------------------

typedef __bf16 bf16_t;
typedef __attribute__((ext_vector_type(16))) __bf16 v16bf;
typedef __attribute__((ext_vector_type(8)))  __bf16 v8bf;
typedef __attribute__((ext_vector_type(8)))  float  v8f;

// ---- WMMA fragment helpers (CDNA5 ISA 7.12.2 layouts) ----------------

// A-matrix 16x32 bf16, row-major source with leading dim `ld`, tile at (0,k0).
// lane<16 : M=lane,    elems0..7 = K k0+0..7,  elems8..15 = K k0+16..23
// lane>=16: M=lane-16, elems0..7 = K k0+8..15, elems8..15 = K k0+24..31
__device__ __forceinline__ v16bf load_a_frag(const bf16_t* base, int ld, int k0) {
  const int lane = threadIdx.x & 31;
  const int m    = lane & 15;
  const int koff = (lane & 16) ? 8 : 0;
  const bf16_t* p = base + (size_t)m * ld + k0 + koff;
  v8bf lo = *(const v8bf*)(p);
  v8bf hi = *(const v8bf*)(p + 16);
  v16bf r;
#pragma unroll
  for (int i = 0; i < 8; ++i) { r[i] = lo[i]; r[i + 8] = hi[i]; }
  return r;
}

// B-matrix 32x16 bf16 where B[k][n] = W[n0+n][k] (W row-major, ld = K stride).
// lane<16 : N=lane,    elems0..7 = K k0+0..7,  elems8..15 = K k0+16..23
// lane>=16: N=lane-16, elems0..7 = K k0+8..15, elems8..15 = K k0+24..31
__device__ __forceinline__ v16bf load_b_frag(const bf16_t* Wn0, int ld, int k0) {
  const int lane = threadIdx.x & 31;
  const int n    = lane & 15;
  const int koff = (lane & 16) ? 8 : 0;
  const bf16_t* p = Wn0 + (size_t)n * ld + k0 + koff;
  v8bf lo = *(const v8bf*)(p);
  v8bf hi = *(const v8bf*)(p + 16);
  v16bf r;
#pragma unroll
  for (int i = 0; i < 8; ++i) { r[i] = lo[i]; r[i + 8] = hi[i]; }
  return r;
}

__device__ __forceinline__ v8f wmma_bf16(v16bf a, v16bf b, v8f c) {
  return __builtin_amdgcn_wmma_f32_16x16x32_bf16(
      /*neg_a=*/false, a, /*neg_b=*/false, b,
      /*c_mod=*/(short)0, c, /*reuse_a=*/false, /*reuse_b=*/false);
}

__device__ __forceinline__ float sigf(float x) { return 1.0f / (1.0f + expf(-x)); }

// ---- small utility kernels -------------------------------------------------

__global__ void cvt_f32_bf16(const float* __restrict__ src, bf16_t* __restrict__ dst, int n) {
  int i = blockIdx.x * blockDim.x + threadIdx.x;
  if (i < n) dst[i] = (bf16_t)src[i];
}

__global__ void copy_f32(const float* __restrict__ src, float* __restrict__ dst, int n) {
  int i = blockIdx.x * blockDim.x + threadIdx.x;
  if (i < n) dst[i] = src[i];
}

// ---- fused LSTM step: both directions, GEMM + gate epilogue ----------------
// Grid: 128 blocks x 256 threads. blockIdx: dir(1b) | m-tile(2b) | hh-chunk(4b)
// Each block: 16 batch rows x 32 hidden cols x 4 gates = 8 WMMA tiles (1/wave),
// K = 512 (x part, gathered embedding) + 512 (h part, recurrent).
__global__ __launch_bounds__(256) void lstm_step(
    int t,
    const int*    __restrict__ sent,     // [64*512] flat token ids
    const float*  __restrict__ emb,      // [32000,512]
    const bf16_t* __restrict__ WiF, const bf16_t* __restrict__ WhF,
    const float*  __restrict__ biF, const float*  __restrict__ bhF,
    const bf16_t* __restrict__ WiB, const bf16_t* __restrict__ WhB,
    const float*  __restrict__ biB, const float*  __restrict__ bhB,
    const bf16_t* __restrict__ h0f, const bf16_t* __restrict__ h0b,
    float* __restrict__ cF, float* __restrict__ cB,
    bf16_t* __restrict__ hsF, bf16_t* __restrict__ hsB) {

  __shared__ bf16_t xa[16][512];       // embedding stripe (bf16)      16 KB
  __shared__ bf16_t ha[16][512];       // previous hidden stripe       16 KB
  __shared__ float  gbuf[4][16][32];   // gate exchange                 8 KB

  const int tid = threadIdx.x;
  const int dir = blockIdx.x >> 6;
  const int rem = blockIdx.x & 63;
  const int mt  = rem >> 4;            // 0..3  (16-row batch tile)
  const int hc  = rem & 15;            // 0..15 (32-col hidden chunk)
  const int m0  = mt * 16;

  const int tt = (dir == 0) ? t : (511 - t);   // time index this dir processes

  const bf16_t* Wi = (dir == 0) ? WiF : WiB;
  const bf16_t* Wh = (dir == 0) ? WhF : WhB;
  const float*  bi = (dir == 0) ? biF : biB;
  const float*  bh = (dir == 0) ? bhF : bhB;
  float*        cD = (dir == 0) ? cF  : cB;
  bf16_t*       hs = (dir == 0) ? hsF : hsB;

  // recurrent input: previous step's row block of hs (double-buffered by time)
  const bf16_t* hprev;
  if (t == 0) {
    hprev = ((dir == 0) ? h0f : h0b) + (size_t)m0 * 512;
  } else {
    const int tprev = (dir == 0) ? (t - 1) : (512 - t);
    hprev = hs + ((size_t)tprev * 64 + m0) * 512;
  }

  // ---- stage A tiles into LDS (shared by all 8 waves of this block) ----
  {
    const int row   = tid >> 4;          // 0..15
    const int cbase = (tid & 15) * 32;   // 0..480
    const int tok   = sent[tt * 64 + m0 + row];
    const float* er = emb + (size_t)tok * 512 + cbase;
#pragma unroll 8
    for (int j = 0; j < 32; ++j) xa[row][cbase + j] = (bf16_t)er[j];
    const bf16_t* hr = hprev + (size_t)row * 512 + cbase;
#pragma unroll 8
    for (int j = 0; j < 32; ++j) ha[row][cbase + j] = hr[j];
  }
  __syncthreads();

  // ---- WMMA: one 16x16 output tile per wave, K = 512(x) + 512(h) ----
  const int wave = tid >> 5;
  const int g    = wave >> 1;            // gate 0..3 (i,f,g,o)
  const int sub  = wave & 1;             // which 16-col half of the 32-col chunk
  const int n0   = g * 512 + hc * 32 + sub * 16;

  const bf16_t* Win0 = Wi + (size_t)n0 * 512;
  const bf16_t* Whn0 = Wh + (size_t)n0 * 512;

  v8f acc = {};
#pragma unroll
  for (int k0 = 0; k0 < 512; k0 += 32)
    acc = wmma_bf16(load_a_frag(&xa[0][0], 512, k0), load_b_frag(Win0, 512, k0), acc);
#pragma unroll
  for (int k0 = 0; k0 < 512; k0 += 32)
    acc = wmma_bf16(load_a_frag(&ha[0][0], 512, k0), load_b_frag(Whn0, 512, k0), acc);

  // scatter accumulator into the gate-exchange buffer (C/D f32 layout)
  {
    const int lane = tid & 31;
    const int nl   = sub * 16 + (lane & 15);
    const int mb   = (lane & 16) ? 8 : 0;
#pragma unroll
    for (int i = 0; i < 8; ++i) gbuf[g][mb + i][nl] = acc[i];
  }
  __syncthreads();

  // ---- gate epilogue: c' = sig(f)*c + sig(i)*tanh(g); h = sig(o)*tanh(c')
#pragma unroll
  for (int p = tid; p < 512; p += 256) {
    const int mi = p >> 5;
    const int hl = p & 31;
    const int hh = hc * 32 + hl;
    const float gi = gbuf[0][mi][hl] + bi[hh]        + bh[hh];
    const float gf = gbuf[1][mi][hl] + bi[512 + hh]  + bh[512 + hh];
    const float gg = gbuf[2][mi][hl] + bi[1024 + hh] + bh[1024 + hh];
    const float go = gbuf[3][mi][hl] + bi[1536 + hh] + bh[1536 + hh];
    const int m = m0 + mi;
    const size_t ci = (size_t)m * 512 + hh;
    const float cold = cD[ci];
    const float cnew = sigf(gf) * cold + sigf(gi) * tanhf(gg);
    const float hnew = sigf(go) * tanhf(cnew);
    cD[ci] = cnew;
    hs[((size_t)tt * 64 + m) * 512 + hh] = (bf16_t)hnew;
  }
}

// ---- tag projection: feats[32768,16] = [hf|hb] @ Wtag^T + b_tag ------------
__global__ __launch_bounds__(256) void tag_proj(
    const bf16_t* __restrict__ hsF, const bf16_t* __restrict__ hsB,
    const bf16_t* __restrict__ Wt,  const float* __restrict__ bt,
    float* __restrict__ feats) {
  const int wave = (blockIdx.x << 3) + (threadIdx.x >> 5);   // 0..2047
  const size_t m0 = (size_t)wave * 16;
  const bf16_t* Af = hsF + m0 * 512;
  const bf16_t* Ab = hsB + m0 * 512;
  v8f acc = {};
#pragma unroll
  for (int k0 = 0; k0 < 512; k0 += 32)
    acc = wmma_bf16(load_a_frag(Af, 512, k0), load_b_frag(Wt, 1024, k0), acc);
#pragma unroll
  for (int k0 = 0; k0 < 512; k0 += 32)
    acc = wmma_bf16(load_a_frag(Ab, 512, k0), load_b_frag(Wt, 1024, 512 + k0), acc);

  const int lane = threadIdx.x & 31;
  const int n    = lane & 15;
  const int mb   = (lane & 16) ? 8 : 0;
  const float bias = bt[n];
#pragma unroll
  for (int i = 0; i < 8; ++i)
    feats[(m0 + mb + i) * 16 + n] = acc[i] + bias;
}

// ---- Viterbi DP + backtrack (single workgroup; T=16, B=64, S=512) ----------
__global__ __launch_bounds__(1024) void viterbi_kernel(
    const float* __restrict__ feats,   // [32768,16]; row(b,s) = b*512+s
    const float* __restrict__ trans,   // [16,16] trans[next][prev]
    unsigned char* __restrict__ bps,   // [512][64][16]
    float* __restrict__ out) {         // [64] scores ++ [64*512] paths
  __shared__ float fv[2][64][16];
  __shared__ float tr[16][16];
  __shared__ float scS[64];
  __shared__ int   blS[64];

  const int tid = threadIdx.x;
  if (tid < 256) tr[tid >> 4][tid & 15] = trans[tid];
  const int b  = tid >> 4;
  const int nt = tid & 15;
  fv[0][b][nt] = (nt == 14) ? 0.0f : -10000.0f;   // START_IDX = 14
  __syncthreads();

  int cur = 0;
  for (int s = 0; s < 512; ++s) {
    float best = -3.4e38f; int bp = 0;
#pragma unroll
    for (int p = 0; p < 16; ++p) {
      const float v = fv[cur][b][p] + tr[nt][p];
      if (v > best) { best = v; bp = p; }            // first-max wins (argmax)
    }
    fv[cur ^ 1][b][nt] = best + feats[((size_t)b * 512 + s) * 16 + nt];
    bps[(size_t)s * 1024 + b * 16 + nt] = (unsigned char)bp;
    cur ^= 1;
    __syncthreads();
  }

  if (nt == 0) {                                     // terminal: + trans[STOP=15]
    float best = -3.4e38f; int bl = 0;
#pragma unroll
    for (int tg = 0; tg < 16; ++tg) {
      const float v = fv[cur][b][tg] + tr[15][tg];
      if (v > best) { best = v; bl = tg; }
    }
    scS[b] = best; blS[b] = bl;
  }
  __syncthreads();

  if (tid < 64) {
    out[tid] = scS[tid];
    int tag = blS[tid];
    for (int s = 511; s >= 0; --s) {
      out[64 + (size_t)tid * 512 + s] = (float)tag;
      tag = (int)bps[(size_t)s * 1024 + tid * 16 + tag];
    }
  }
}

// ---------------------------------------------------------------------------

extern "C" void kernel_launch(void* const* d_in, const int* in_sizes, int n_in,
                              void* d_out, int out_size, void* d_ws, size_t ws_size,
                              hipStream_t stream) {
  (void)in_sizes; (void)n_in; (void)out_size; (void)ws_size;

  const int*   sent  = (const int*)  d_in[0];
  const float* emb   = (const float*)d_in[1];
  const float* Wi_f  = (const float*)d_in[2];
  const float* Wh_f  = (const float*)d_in[3];
  const float* bi_f  = (const float*)d_in[4];
  const float* bh_f  = (const float*)d_in[5];
  const float* Wi_b  = (const float*)d_in[6];
  const float* Wh_b  = (const float*)d_in[7];
  const float* bi_b  = (const float*)d_in[8];
  const float* bh_b  = (const float*)d_in[9];
  const float* W_tag = (const float*)d_in[10];
  const float* b_tag = (const float*)d_in[11];
  const float* trans = (const float*)d_in[12];
  const float* h0    = (const float*)d_in[13];
  const float* c0    = (const float*)d_in[14];
  float* out = (float*)d_out;

  // workspace carve (~79 MB total)
  char* ws = (char*)d_ws;
  size_t off = 0;
  auto carve = [&](size_t bytes) { void* p = ws + off; off = (off + bytes + 255) & ~(size_t)255; return p; };

  const size_t NW = 2048 * 512;                 // per weight matrix
  bf16_t* wWiF = (bf16_t*)carve(NW * 2);
  bf16_t* wWhF = (bf16_t*)carve(NW * 2);
  bf16_t* wWiB = (bf16_t*)carve(NW * 2);
  bf16_t* wWhB = (bf16_t*)carve(NW * 2);
  bf16_t* wWt  = (bf16_t*)carve(16 * 1024 * 2);
  bf16_t* h0f  = (bf16_t*)carve(64 * 512 * 2);
  bf16_t* h0b  = (bf16_t*)carve(64 * 512 * 2);
  float*  cF   = (float*) carve(64 * 512 * 4);
  float*  cB   = (float*) carve(64 * 512 * 4);
  bf16_t* hsF  = (bf16_t*)carve((size_t)32768 * 512 * 2);
  bf16_t* hsB  = (bf16_t*)carve((size_t)32768 * 512 * 2);
  float*  feats= (float*) carve((size_t)32768 * 16 * 4);
  unsigned char* bps = (unsigned char*)carve((size_t)512 * 64 * 16);

  // 1) convert weights / initial state
  const int T256 = 256;
  cvt_f32_bf16<<<(int)((NW + 255) / 256), T256, 0, stream>>>(Wi_f, wWiF, (int)NW);
  cvt_f32_bf16<<<(int)((NW + 255) / 256), T256, 0, stream>>>(Wh_f, wWhF, (int)NW);
  cvt_f32_bf16<<<(int)((NW + 255) / 256), T256, 0, stream>>>(Wi_b, wWiB, (int)NW);
  cvt_f32_bf16<<<(int)((NW + 255) / 256), T256, 0, stream>>>(Wh_b, wWhB, (int)NW);
  cvt_f32_bf16<<<(16 * 1024 + 255) / 256, T256, 0, stream>>>(W_tag, wWt, 16 * 1024);
  cvt_f32_bf16<<<(32768 + 255) / 256, T256, 0, stream>>>(h0,          h0f, 32768);
  cvt_f32_bf16<<<(32768 + 255) / 256, T256, 0, stream>>>(h0 + 32768,  h0b, 32768);
  copy_f32    <<<(32768 + 255) / 256, T256, 0, stream>>>(c0,          cF,  32768);
  copy_f32    <<<(32768 + 255) / 256, T256, 0, stream>>>(c0 + 32768,  cB,  32768);

  // 2) sequential LSTM, both directions fused per step
  for (int t = 0; t < 512; ++t) {
    lstm_step<<<128, 256, 0, stream>>>(t, sent, emb,
                                       wWiF, wWhF, bi_f, bh_f,
                                       wWiB, wWhB, bi_b, bh_b,
                                       h0f, h0b, cF, cB, hsF, hsB);
  }

  // 3) tag projection (WMMA, N = 16 tags)
  tag_proj<<<256, 256, 0, stream>>>(hsF, hsB, wWt, b_tag, feats);

  // 4) Viterbi DP + backtrack
  viterbi_kernel<<<1, 1024, 0, stream>>>(feats, trans, bps, out);
}
